// PointRCNNBlock_65335042507282
// MI455X (gfx1250) — compile-verified, matrix operand order checked by reading, not run on hardware
//
#include <hip/hip_runtime.h>
#include <hip/hip_bf16.h>
#include <math.h>

#define DEVI static __device__ __forceinline__

typedef __attribute__((ext_vector_type(16))) __bf16 bf16x16;
typedef __attribute__((ext_vector_type(8)))  float  f32x8;

constexpr int CD   = 128;           // hidden dim
constexpr int BSZ  = 4;
constexpr int NT   = 1024;
constexpr int BN   = BSZ * NT;      // 4096 tokens
constexpr int RR   = 49;
constexpr int NOUT = CD * RR;       // 6272

// ---------------- WMMA fragment helpers (16x16x32 bf16, wave32) -------------
// 16-bit A (MxK=16x32): lane l: M=l%16, half h=l/16; element e -> K = (e/8)*16 + h*8 + (e%8)
DEVI int kmap(int e, int h) { return ((e >> 3) << 4) + (h << 3) + (e & 7); }

DEVI f32x8 wmma_bf16(bf16x16 a, bf16x16 b, f32x8 c) {
  return __builtin_amdgcn_wmma_f32_16x16x32_bf16(false, a, false, b, (short)0, c,
                                                 false, false);
}

// A-fragment from LDS tile, row-major bf16, leading dim ld
DEVI bf16x16 fragA(const __bf16* t, int ld, int m0, int k0) {
  int lane = threadIdx.x & 31;
  int m = m0 + (lane & 15), h = lane >> 4;
  bf16x16 a;
#pragma unroll
  for (int e = 0; e < 16; ++e) a[e] = t[m * ld + k0 + kmap(e, h)];
  return a;
}

// A-fragment with row predicate (zero-pad rows >= rows)
DEVI bf16x16 fragA_pred(const __bf16* t, int ld, int m0, int k0, int rows) {
  int lane = threadIdx.x & 31;
  int m = m0 + (lane & 15), h = lane >> 4;
  bf16x16 a;
#pragma unroll
  for (int e = 0; e < 16; ++e)
    a[e] = (m < rows) ? t[m * ld + k0 + kmap(e, h)] : (__bf16)0.0f;
  return a;
}

// B-fragment (KxN) built from a row-major weight W (N x K): B[k][n] = W[n][k]
DEVI bf16x16 fragB_w(const __bf16* W, int ldw, int n0, int k0) {
  int lane = threadIdx.x & 31;
  int n = n0 + (lane & 15), h = lane >> 4;
  bf16x16 b;
#pragma unroll
  for (int e = 0; e < 16; ++e) b[e] = W[(size_t)n * ldw + k0 + kmap(e, h)];
  return b;
}

// B-fragment from [K][N] bf16 LDS tile
DEVI bf16x16 fragB_kn(const __bf16* t, int ld, int n0, int k0) {
  int lane = threadIdx.x & 31;
  int n = n0 + (lane & 15), h = lane >> 4;
  bf16x16 b;
#pragma unroll
  for (int e = 0; e < 16; ++e) b[e] = t[(k0 + kmap(e, h)) * ld + n];
  return b;
}

// -------- reductions (wave32; 16-lane half-group for C-fragment rows) -------
DEVI float hmax16(float x) {
  x = fmaxf(x, __shfl_xor(x, 1, 32));
  x = fmaxf(x, __shfl_xor(x, 2, 32));
  x = fmaxf(x, __shfl_xor(x, 4, 32));
  x = fmaxf(x, __shfl_xor(x, 8, 32));
  return x;
}
DEVI float hsum16(float x) {
  x += __shfl_xor(x, 1, 32);
  x += __shfl_xor(x, 2, 32);
  x += __shfl_xor(x, 4, 32);
  x += __shfl_xor(x, 8, 32);
  return x;
}

// wave-cooperative LayerNorm over a 128-float LDS row (eps 1e-5)
DEVI void ln_row(float* row, const float* g, const float* b) {
  int lane = threadIdx.x & 31;
  float s0 = 0.f, s1 = 0.f;
  for (int i = lane; i < 128; i += 32) { float v = row[i]; s0 += v; s1 += v * v; }
  for (int s = 1; s < 32; s <<= 1) { s0 += __shfl_xor(s0, s, 32); s1 += __shfl_xor(s1, s, 32); }
  float mean = s0 * (1.0f / 128.0f);
  float var  = s1 * (1.0f / 128.0f) - mean * mean;
  float inv  = rsqrtf(var + 1e-5f);
  for (int i = lane; i < 128; i += 32) row[i] = (row[i] - mean) * inv * g[i] + b[i];
}

// LayerNorm + ReLU in place on a bf16 LDS row of 128 (stats in f32)
DEVI void ln_relu_row_bf16(__bf16* row, const float* g, const float* b) {
  int lane = threadIdx.x & 31;
  float s0 = 0.f, s1 = 0.f;
  for (int i = lane; i < 128; i += 32) { float v = (float)row[i]; s0 += v; s1 += v * v; }
  for (int s = 1; s < 32; s <<= 1) { s0 += __shfl_xor(s0, s, 32); s1 += __shfl_xor(s1, s, 32); }
  float mean = s0 * (1.0f / 128.0f);
  float var  = s1 * (1.0f / 128.0f) - mean * mean;
  float inv  = rsqrtf(var + 1e-5f);
  for (int i = lane; i < 128; i += 32) {
    float v = ((float)row[i] - mean) * inv * g[i] + b[i];
    row[i] = (__bf16)fmaxf(v, 0.0f);
  }
}

// ------------------------------ prep kernels --------------------------------
__global__ void k_cvt(const float* __restrict__ src, __bf16* __restrict__ dst, int n) {
  int i = blockIdx.x * blockDim.x + threadIdx.x;
  if (i < n) dst[i] = (__bf16)src[i];
}

// query_feat (B,C,N) -> xT (B*N, C) f32 + xb bf16
__global__ void k_trx(const float* __restrict__ qf, float* __restrict__ xT,
                      __bf16* __restrict__ xb) {
  int i = blockIdx.x * blockDim.x + threadIdx.x;   // over B*C*N = 524288
  int n = i & 1023, c = (i >> 10) & 127, b = i >> 17;
  float v = qf[i];
  int ro = (b << 10) + n;
  xT[(size_t)ro * 128 + c] = v;
  xb[(size_t)ro * 128 + c] = (__bf16)v;
}

// ---------------------- QKV projection: qkv = x @ Wqkv^T + b ----------------
__global__ __launch_bounds__(256) void k_qkv(const __bf16* __restrict__ xb,
                                             const __bf16* __restrict__ Wb,
                                             const float* __restrict__ bias,
                                             __bf16* __restrict__ qkvb) {
  __shared__ __bf16 aa[16 * 130];
  int r0 = blockIdx.x * 16;
  for (int i = threadIdx.x; i < 16 * 128; i += 256) {
    int mm = i >> 7, k = i & 127;
    aa[mm * 130 + k] = xb[(size_t)(r0 + mm) * 128 + k];
  }
  __syncthreads();
  int w = threadIdx.x >> 5, lane = threadIdx.x & 31;
  int n = lane & 15, h = lane >> 4;
  for (int j = 0; j < 3; ++j) {
    int n0 = (w + 8 * j) * 16;
    f32x8 c = {};
#pragma unroll
    for (int k0 = 0; k0 < 128; k0 += 32)
      c = wmma_bf16(fragA(aa, 130, 0, k0), fragB_w(Wb, 128, n0, k0), c);
    int col = n0 + n;
#pragma unroll
    for (int v = 0; v < 8; ++v)
      qkvb[(size_t)(r0 + v + 8 * h) * 384 + col] = (__bf16)(c[v] + bias[col]);
  }
}

// ------------- attention: flash-style, one wave per 16-query tile -----------
__global__ __launch_bounds__(256) void k_attn(const __bf16* __restrict__ qkvb,
                                              float* __restrict__ attnO) {
  __shared__ float patch[8][16 * 17];
  int blk = blockIdx.x;                 // B*H*8 = 256 blocks
  int b  = blk >> 6;
  int hd = (blk >> 3) & 7;
  int qg = blk & 7;
  int w = threadIdx.x >> 5, lane = threadIdx.x & 31;
  int m = lane & 15, h = lane >> 4;
  int rb = b << 10;
  int q0 = (qg * 8 + w) * 16;
  const __bf16* Qp = qkvb + (size_t)rb * 384 + hd * 16;
  const __bf16* Kp = Qp + 128;
  const __bf16* Vp = Qp + 256;

  bf16x16 qa;                           // A: 16 queries x K=32 (d=16, zero pad)
#pragma unroll
  for (int e = 0; e < 16; ++e) {
    int k = kmap(e, h);
    qa[e] = (k < 16) ? Qp[(size_t)(q0 + m) * 384 + k] : (__bf16)0.0f;
  }
  f32x8 o = {};
  float mrow[8], lrow[8];
#pragma unroll
  for (int v = 0; v < 8; ++v) { mrow[v] = -3.0e38f; lrow[v] = 0.0f; }
  float* pt = patch[w];

  for (int kt = 0; kt < 64; ++kt) {
    int kb = kt * 16;
    bf16x16 kf;                         // B[k=d][n=key]
#pragma unroll
    for (int e = 0; e < 16; ++e) {
      int k = kmap(e, h);
      kf[e] = (k < 16) ? Kp[(size_t)(kb + m) * 384 + k] : (__bf16)0.0f;
    }
    f32x8 z = {};
    f32x8 s = wmma_bf16(qa, kf, z);     // S tile 16x16
    __syncthreads();                    // previous-iter patch reads done
#pragma unroll
    for (int v = 0; v < 8; ++v) {
      float sv = s[v] * 0.25f;          // 1/sqrt(16)
      float nm = fmaxf(mrow[v], hmax16(sv));
      float corr = __expf(mrow[v] - nm);
      float p = __expf(sv - nm);
      lrow[v] = lrow[v] * corr + hsum16(p);
      o[v] *= corr;
      mrow[v] = nm;
      pt[(v + 8 * h) * 17 + m] = p;     // C-layout -> LDS
    }
    __syncthreads();
    bf16x16 pa;                         // P as A: 16q x K=32 (16 keys + pad)
#pragma unroll
    for (int e = 0; e < 16; ++e) {
      int k = kmap(e, h);
      pa[e] = (k < 16) ? (__bf16)pt[m * 17 + k] : (__bf16)0.0f;
    }
    bf16x16 vf;                         // B[k=key][n=d]
#pragma unroll
    for (int e = 0; e < 16; ++e) {
      int k = kmap(e, h);
      vf[e] = (k < 16) ? Vp[(size_t)(kb + k) * 384 + m] : (__bf16)0.0f;
    }
    o = wmma_bf16(pa, vf, o);
  }
#pragma unroll
  for (int v = 0; v < 8; ++v) {
    int q = q0 + v + 8 * h;
    attnO[(size_t)(rb + q) * 128 + hd * 16 + m] = o[v] / lrow[v];
  }
}

// ---------------- Wo projection + residual + LayerNorm1 ---------------------
__global__ __launch_bounds__(256) void k_wo(const float* __restrict__ attnO,
                                            const __bf16* __restrict__ Wob,
                                            const float* __restrict__ bo,
                                            const float* __restrict__ xT,
                                            const float* __restrict__ g1,
                                            const float* __restrict__ b1,
                                            float* __restrict__ x1,
                                            __bf16* __restrict__ x1b) {
  __shared__ __bf16 aa[16 * 130];
  __shared__ float  yt[16 * 132];
  int r0 = blockIdx.x * 16;
  for (int i = threadIdx.x; i < 16 * 128; i += 256) {
    int mm = i >> 7, k = i & 127;
    aa[mm * 130 + k] = (__bf16)attnO[(size_t)(r0 + mm) * 128 + k];
  }
  __syncthreads();
  int w = threadIdx.x >> 5, lane = threadIdx.x & 31;
  int n0 = w * 16;
  f32x8 c = {};
#pragma unroll
  for (int k0 = 0; k0 < 128; k0 += 32)
    c = wmma_bf16(fragA(aa, 130, 0, k0), fragB_w(Wob, 128, n0, k0), c);
  int n = n0 + (lane & 15), h = lane >> 4;
#pragma unroll
  for (int v = 0; v < 8; ++v) yt[(v + 8 * h) * 132 + n] = c[v] + bo[n];
  __syncthreads();
  for (int rr = w; rr < 16; rr += 8) {
    float* row = yt + rr * 132;
    const float* xr = xT + (size_t)(r0 + rr) * 128;
    for (int i = lane; i < 128; i += 32) row[i] += xr[i];
    ln_row(row, g1, b1);
    for (int i = lane; i < 128; i += 32) {
      float v = row[i];
      x1 [(size_t)(r0 + rr) * 128 + i] = v;
      x1b[(size_t)(r0 + rr) * 128 + i] = (__bf16)v;
    }
  }
}

// ------- dynamic conv, 8 tokens per workgroup, all-WMMA, big-LDS fused ------
// Dynamic LDS layout (bytes):
//   Fb1 : 8 * 49*132 bf16 = 103488   (conv1 raw -> f1)
//   Fb2 : 8 * 49*132 bf16 = 103488   (roi staging -> conv2 raw -> f2)
//   Pb  : 64*130     bf16 = 16640    (param tile for one c-half)
//   Xg  : 16*130     bf16 = 4160     (x1 tokens, rows 8..15 zero)
//   xtok: 8*128      f32  = 4096
//   yb  : 8*132      f32  = 4224
constexpr int FB_STRIDE = 132;
constexpr int FB_TOK    = RR * FB_STRIDE;        // 6468
constexpr unsigned OFF_FB1  = 0;
constexpr unsigned OFF_FB2  = 103488;
constexpr unsigned OFF_PB   = 206976;
constexpr unsigned OFF_XG   = 223616;
constexpr unsigned OFF_XTOK = 227776;
constexpr unsigned OFF_YB   = 231872;
constexpr unsigned SMEM_DYN = 236096;

__global__ __launch_bounds__(256) void k_dyn(
    const float* __restrict__ roi, const float* __restrict__ x1,
    const __bf16* __restrict__ x1b,
    const __bf16* __restrict__ dynWb, const float* __restrict__ dynb,
    const __bf16* __restrict__ outWb, const float* __restrict__ outb,
    const float* __restrict__ g_dn1, const float* __restrict__ b_dn1,
    const float* __restrict__ g_dn2, const float* __restrict__ b_dn2,
    const float* __restrict__ g_dn3, const float* __restrict__ b_dn3,
    const float* __restrict__ g_ln2, const float* __restrict__ b_ln2,
    float* __restrict__ x2, __bf16* __restrict__ x2b) {
  extern __shared__ char smem[];
  __bf16* Fb1  = (__bf16*)(smem + OFF_FB1);
  __bf16* Fb2  = (__bf16*)(smem + OFF_FB2);
  __bf16* Pb   = (__bf16*)(smem + OFF_PB);
  __bf16* Xg   = (__bf16*)(smem + OFF_XG);
  float*  xtok = (float*)(smem + OFF_XTOK);
  float*  yb   = (float*)(smem + OFF_YB);

  int t0 = blockIdx.x * 8;
  int tid = threadIdx.x;
  int w = tid >> 5, lane = tid & 31;
  int h = lane >> 4;

  // stage x1 tokens (bf16 A rows, rows 8..15 zero) + f32 copy for residual
  for (int i = tid; i < 16 * 130; i += 256) {
    int m = i / 130, k = i - m * 130;
    Xg[i] = (m < 8 && k < 128) ? x1b[(size_t)(t0 + m) * 128 + k] : (__bf16)0.0f;
  }
  for (int i = tid; i < 8 * 128; i += 256)
    xtok[i] = x1[(size_t)t0 * 128 + i];
  // stage roi -> Fb2 (bf16)
  for (int i = tid; i < 8 * RR * 128; i += 256) {
    int t = i / (RR * 128);
    int rem = i - t * (RR * 128);
    int r = rem >> 7, c = rem & 127;
    int tok = t0 + t;
    int b4 = tok >> 10, n = tok & 1023;
    Fb2[t * FB_TOK + r * FB_STRIDE + c] =
        (__bf16)roi[((size_t)(b4 * RR + r) * 1024 + n) * 128 + c];
  }
  __syncthreads();

  for (int pass = 0; pass < 2; ++pass) {
    int rowoff = pass * 16384;
    const __bf16* Asrc = (pass == 0) ? Fb2 : Fb1;
    __bf16* Fdst = (pass == 0) ? Fb1 : Fb2;

    for (int dt = 0; dt < 8; ++dt) {
      int d0 = dt * 16;
      f32x8 cacc[4] = {{}, {}, {}, {}};
      for (int ch = 0; ch < 2; ++ch) {
        // ---- parameter GEMM: P[t, c, j] = X(16x128) @ dynW^T, 8 c per wave
#pragma unroll
        for (int ci = 0; ci < 8; ++ci) {
          int c = ch * 64 + w * 8 + ci;
          const __bf16* wrow = dynWb + (size_t)(rowoff + c * 128 + d0) * 128;
          f32x8 pc = {};
#pragma unroll
          for (int k0 = 0; k0 < 128; k0 += 32)
            pc = wmma_bf16(fragA(Xg, 130, 0, k0), fragB_w(wrow, 128, 0, k0), pc);
          if (h == 0) {
            int j = lane & 15;
            float bias = dynb[rowoff + c * 128 + d0 + j];
#pragma unroll
            for (int v = 0; v < 8; ++v)
              Pb[(c - ch * 64) * 130 + v * 16 + j] = (__bf16)(pc[v] + bias);
          }
        }
        __syncthreads();
        // ---- conv partial: token = wave; F(49x16) += A(49x64) @ P(64x16)
        const __bf16* As = Asrc + w * FB_TOK;
#pragma unroll
        for (int mt = 0; mt < 4; ++mt) {
#pragma unroll
          for (int kk = 0; kk < 2; ++kk) {
            int crel = kk * 32;
            bf16x16 a = fragA_pred(As, FB_STRIDE, mt * 16, ch * 64 + crel, RR);
            bf16x16 bfr = fragB_kn(Pb + w * 16, 130, 0, crel);
            cacc[mt] = wmma_bf16(a, bfr, cacc[mt]);
          }
        }
        __syncthreads();
      }
      // store this d-tile (pre-LN raw) for wave's token
      {
        int n = lane & 15;
        __bf16* Fd = Fdst + w * FB_TOK;
#pragma unroll
        for (int mt = 0; mt < 4; ++mt)
#pragma unroll
          for (int v = 0; v < 8; ++v) {
            int m = mt * 16 + v + 8 * h;
            if (m < RR) Fd[m * FB_STRIDE + d0 + n] = (__bf16)cacc[mt][v];
          }
      }
    }
    __syncthreads();
    // LayerNorm + ReLU on every (token, r) row of Fdst
    const float* g = pass ? g_dn2 : g_dn1;
    const float* bb = pass ? b_dn2 : b_dn1;
    for (int idx = w; idx < 8 * RR; idx += 8) {
      int t = idx / RR, r = idx - t * RR;
      ln_relu_row_bf16(Fdst + t * FB_TOK + r * FB_STRIDE, g, bb);
    }
    __syncthreads();
  }

  // ---- out projection: y(8tok x 128) = f2(8 x 6272) @ outW^T, WMMA, K=6272
  {
    int n0 = w * 16;
    f32x8 oc = {};
    for (int ks = 0; ks < 196; ++ks) {
      int kq = ks * 32;
      bf16x16 a;
      int tt = lane & 15;
#pragma unroll
      for (int e = 0; e < 16; ++e) {
        int k = kq + kmap(e, h);
        int r = k >> 7, c = k & 127;
        a[e] = (tt < 8) ? Fb2[tt * FB_TOK + r * FB_STRIDE + c] : (__bf16)0.0f;
      }
      bf16x16 bfr = fragB_w(outWb, NOUT, n0, kq);
      oc = wmma_bf16(a, bfr, oc);
    }
    if (h == 0) {
      int n = n0 + (lane & 15);
#pragma unroll
      for (int v = 0; v < 8; ++v) yb[v * FB_STRIDE + n] = oc[v] + outb[n];
    }
  }
  __syncthreads();

  // ---- per-token epilogue: dn3 LN -> ReLU -> +x1 -> ln2 -> x2
  {
    float* row = yb + w * FB_STRIDE;
    ln_row(row, g_dn3, b_dn3);
    for (int i = lane; i < 128; i += 32)
      row[i] = xtok[w * 128 + i] + fmaxf(row[i], 0.0f);
    ln_row(row, g_ln2, b_ln2);
    for (int i = lane; i < 128; i += 32) {
      float v = row[i];
      size_t gi = (size_t)(t0 + w) * 128 + i;
      x2[gi] = v;
      x2b[gi] = (__bf16)v;
    }
  }
}

// ---------------------------- FFN1 (exact GELU) -----------------------------
__global__ __launch_bounds__(256) void k_ff1(const __bf16* __restrict__ x2b,
                                             const __bf16* __restrict__ W1b,
                                             const float* __restrict__ bias,
                                             __bf16* __restrict__ hb) {
  __shared__ __bf16 aa[16 * 130];
  int r0 = blockIdx.x * 16;
  for (int i = threadIdx.x; i < 16 * 128; i += 256) {
    int mm = i >> 7, k = i & 127;
    aa[mm * 130 + k] = x2b[(size_t)(r0 + mm) * 128 + k];
  }
  __syncthreads();
  int w = threadIdx.x >> 5, lane = threadIdx.x & 31;
  int n = lane & 15, h = lane >> 4;
  for (int j = 0; j < 4; ++j) {
    int n0 = (w + 8 * j) * 16;
    f32x8 c = {};
#pragma unroll
    for (int k0 = 0; k0 < 128; k0 += 32)
      c = wmma_bf16(fragA(aa, 130, 0, k0), fragB_w(W1b, 128, n0, k0), c);
    int col = n0 + n;
#pragma unroll
    for (int v = 0; v < 8; ++v) {
      float x = c[v] + bias[col];
      float gel = 0.5f * x * (1.0f + erff(x * 0.70710678118654752f));
      hb[(size_t)(r0 + v + 8 * h) * 512 + col] = (__bf16)gel;
    }
  }
}

// ------------------ FFN2 + residual + LN3 + output transpose ----------------
__global__ __launch_bounds__(256) void k_ff2(const __bf16* __restrict__ hb,
                                             const __bf16* __restrict__ W2b,
                                             const float* __restrict__ bias,
                                             const float* __restrict__ x2,
                                             const float* __restrict__ g3,
                                             const float* __restrict__ b3,
                                             float* __restrict__ out) {
  __shared__ __bf16 aa[16 * 514];
  __shared__ float  yt[16 * 132];
  int r0 = blockIdx.x * 16;
  for (int i = threadIdx.x; i < 16 * 512; i += 256) {
    int mm = i >> 9, k = i & 511;
    aa[mm * 514 + k] = hb[(size_t)(r0 + mm) * 512 + k];
  }
  __syncthreads();
  int w = threadIdx.x >> 5, lane = threadIdx.x & 31;
  int n0 = w * 16;
  f32x8 c = {};
  for (int k0 = 0; k0 < 512; k0 += 32)
    c = wmma_bf16(fragA(aa, 514, 0, k0), fragB_w(W2b, 512, n0, k0), c);
  int n = n0 + (lane & 15), h = lane >> 4;
#pragma unroll
  for (int v = 0; v < 8; ++v) yt[(v + 8 * h) * 132 + n] = c[v] + bias[n];
  __syncthreads();
  for (int rr = w; rr < 16; rr += 8) {
    float* row = yt + rr * 132;
    int grow = r0 + rr;
    const float* xr = x2 + (size_t)grow * 128;
    for (int i = lane; i < 128; i += 32) row[i] += xr[i];
    ln_row(row, g3, b3);
    int bb = grow >> 10, nn = grow & 1023;
    for (int i = lane; i < 128; i += 32)
      out[(size_t)(bb * 128 + i) * 1024 + nn] = row[i];
  }
}

// ------------------------------- host side ----------------------------------
extern "C" void kernel_launch(void* const* d_in, const int* in_sizes, int n_in,
                              void* d_out, int out_size, void* d_ws, size_t ws_size,
                              hipStream_t stream) {
  (void)in_sizes; (void)n_in; (void)out_size; (void)ws_size;
  const float* query_feat = (const float*)d_in[0];
  const float* roi        = (const float*)d_in[1];
  const float* Wqkv = (const float*)d_in[2];  const float* bqkv = (const float*)d_in[3];
  const float* Wo   = (const float*)d_in[4];  const float* bo   = (const float*)d_in[5];
  const float* ln1g = (const float*)d_in[6];  const float* ln1b = (const float*)d_in[7];
  const float* dynW = (const float*)d_in[8];  const float* dynb = (const float*)d_in[9];
  const float* dn1g = (const float*)d_in[10]; const float* dn1b = (const float*)d_in[11];
  const float* dn2g = (const float*)d_in[12]; const float* dn2b = (const float*)d_in[13];
  const float* outW = (const float*)d_in[14]; const float* outb = (const float*)d_in[15];
  const float* dn3g = (const float*)d_in[16]; const float* dn3b = (const float*)d_in[17];
  const float* ln2g = (const float*)d_in[18]; const float* ln2b = (const float*)d_in[19];
  const float* ff1W = (const float*)d_in[20]; const float* ff1b = (const float*)d_in[21];
  const float* ff2W = (const float*)d_in[22]; const float* ff2b = (const float*)d_in[23];
  const float* ln3g = (const float*)d_in[24]; const float* ln3b = (const float*)d_in[25];

  char* p = (char*)d_ws;
  auto alloc = [&](size_t bytes) -> void* {
    void* r = (void*)p; p += (bytes + 255) & ~((size_t)255); return r;
  };
  float*  xT    = (float*) alloc((size_t)BN * 128 * 4);
  __bf16* xb    = (__bf16*)alloc((size_t)BN * 128 * 2);
  __bf16* Wqkvb = (__bf16*)alloc((size_t)384 * 128 * 2);
  __bf16* Wob   = (__bf16*)alloc((size_t)128 * 128 * 2);
  __bf16* dynWb = (__bf16*)alloc((size_t)32768 * 128 * 2);
  __bf16* outWb = (__bf16*)alloc((size_t)128 * NOUT * 2);
  __bf16* ff1Wb = (__bf16*)alloc((size_t)512 * 128 * 2);
  __bf16* ff2Wb = (__bf16*)alloc((size_t)128 * 512 * 2);
  __bf16* qkvb  = (__bf16*)alloc((size_t)BN * 384 * 2);
  float*  attnO = (float*) alloc((size_t)BN * 128 * 4);
  float*  x1    = (float*) alloc((size_t)BN * 128 * 4);
  __bf16* x1b   = (__bf16*)alloc((size_t)BN * 128 * 2);
  float*  x2    = (float*) alloc((size_t)BN * 128 * 4);
  __bf16* x2b   = (__bf16*)alloc((size_t)BN * 128 * 2);
  __bf16* hb    = (__bf16*)alloc((size_t)BN * 512 * 2);

  k_cvt<<<(384 * 128 + 255) / 256, 256, 0, stream>>>(Wqkv, Wqkvb, 384 * 128);
  k_cvt<<<(128 * 128 + 255) / 256, 256, 0, stream>>>(Wo, Wob, 128 * 128);
  k_cvt<<<(32768 * 128 + 255) / 256, 256, 0, stream>>>(dynW, dynWb, 32768 * 128);
  k_cvt<<<(128 * NOUT + 255) / 256, 256, 0, stream>>>(outW, outWb, 128 * NOUT);
  k_cvt<<<(512 * 128 + 255) / 256, 256, 0, stream>>>(ff1W, ff1Wb, 512 * 128);
  k_cvt<<<(128 * 512 + 255) / 256, 256, 0, stream>>>(ff2W, ff2Wb, 128 * 512);
  k_trx<<<(BSZ * 128 * 1024) / 256, 256, 0, stream>>>(query_feat, xT, xb);

  k_qkv<<<BN / 16, 256, 0, stream>>>(xb, Wqkvb, bqkv, qkvb);
  k_attn<<<BSZ * 8 * 8, 256, 0, stream>>>(qkvb, attnO);
  k_wo<<<BN / 16, 256, 0, stream>>>(attnO, Wob, bo, xT, ln1g, ln1b, x1, x1b);
  k_dyn<<<BN / 8, 256, SMEM_DYN, stream>>>(roi, x1, x1b, dynWb, dynb, outWb, outb,
                                           dn1g, dn1b, dn2g, dn2b, dn3g, dn3b,
                                           ln2g, ln2b, x2, x2b);
  k_ff1<<<BN / 16, 256, 0, stream>>>(x2b, ff1Wb, ff1b, hb);
  k_ff2<<<BN / 16, 256, 0, stream>>>(hb, ff2Wb, ff2b, x2, ln3g, ln3b, (float*)d_out);
}
